// GlobalInteractor_29575144800266
// MI455X (gfx1250) — compile-verified, matrix operand order checked by reading, not run on hardware
//
#include <hip/hip_runtime.h>
#include <math.h>

#define NN 50000
#define NE 1000000
#define NLAYER 3

typedef unsigned short u16;
typedef __attribute__((ext_vector_type(8)))  u16    v8u;
typedef __attribute__((ext_vector_type(16))) u16    v16u;
typedef __attribute__((ext_vector_type(16))) __bf16 v16bf;
typedef __attribute__((ext_vector_type(8)))  float  v8f;
typedef __attribute__((ext_vector_type(4)))  float  v4f;

// ---------------- parameter-tree offsets (jax tree_flatten, sorted keys) ---
#define L_FC1_B 0
#define L_FC1_W 256
#define L_FC2_B 16640
#define L_FC2_W 16704
#define L_HH_B 33088
#define L_HH_W 33152
#define L_IH_B 37248
#define L_IH_W 37312
#define L_KE_B 41408
#define L_KE_W 41472
#define L_KN_B 45568
#define L_KN_W 45632
#define L_N1_B 49728
#define L_N1_G 49792
#define L_N2_B 49856
#define L_N2_G 49920
#define L_OUT_B 49984
#define L_OUT_W 50048
#define L_Q_B 54144
#define L_Q_W 54208
#define L_SELF_B 58304
#define L_SELF_W 58368
#define L_VE_B 62464
#define L_VE_W 62528
#define L_VN_B 66624
#define L_VN_W 66688
#define LAYER_SZ 70784
#define P_NORM_B 212352
#define P_NORM_G 212416
#define P_PROJ_B 212480
#define P_PROJ_W 212864
#define P_REL 237440
#define R_AL_B (P_REL + 0)
#define R_AL_W (P_REL + 64)
#define R_AN1_B (P_REL + 4160)
#define R_AN1_G (P_REL + 4224)
#define R_AN2_B (P_REL + 4288)
#define R_AN2_G (P_REL + 4352)
#define R_B0_L1_B (P_REL + 4416)
#define R_B0_L1_W (P_REL + 4480)
#define R_B0_L2_B (P_REL + 4608)
#define R_B0_L2_W (P_REL + 4672)
#define R_B0_N1_B (P_REL + 8768)
#define R_B0_N1_G (P_REL + 8832)
#define R_B1_L1_B (P_REL + 8896)
#define R_B1_L1_W (P_REL + 8960)
#define R_B1_L2_B (P_REL + 9088)
#define R_B1_L2_W (P_REL + 9152)
#define R_B1_N1_B (P_REL + 13248)
#define R_B1_N1_G (P_REL + 13312)

// ---------------- helpers ---------------------------------------------------
__device__ __forceinline__ u16 f2bf(float x) {
  unsigned u = __float_as_uint(x);
  unsigned r = u + 0x7FFFu + ((u >> 16) & 1u);
  return (u16)(r >> 16);
}

// A fragment: lane%16 = M row; per half, 16 bf16 live in two contiguous 8-elem
// runs -> two ds_load_b128 (stride 72 u16 keeps every run 16B aligned).
__device__ __forceinline__ v16bf ld_a(const u16* l, int row0, int k0, int stride) {
  int lane = threadIdx.x & 31;
  const u16* p = l + (row0 + (lane & 15)) * stride + k0 + ((lane >> 4) << 3);
  v8u lo = *(const v8u*)p;
  v8u hi = *(const v8u*)(p + 16);
  v16u u = __builtin_shufflevector(lo, hi, 0, 1, 2, 3, 4, 5, 6, 7,
                                   8, 9, 10, 11, 12, 13, 14, 15);
  return __builtin_bit_cast(v16bf, u);
}

// B fragment from LDS staged in fragment-swizzled order:
// tile t = kt*ntiles + nt occupies 32 lanes x 16 contiguous u16.
__device__ __forceinline__ v16bf ld_b_sw(const u16* l, int kt, int nt, int ntiles) {
  const u16* p = l + ((((kt * ntiles + nt) << 5) | (threadIdx.x & 31)) << 4);
  v8u lo = *(const v8u*)p;
  v8u hi = *(const v8u*)(p + 8);
  v16u u = __builtin_shufflevector(lo, hi, 0, 1, 2, 3, 4, 5, 6, 7,
                                   8, 9, 10, 11, 12, 13, 14, 15);
  return __builtin_bit_cast(v16bf, u);
}

// stage W (f32, row-major, srcStride cols, starting col c0) -> LDS bf16,
// fragment-swizzled; each thread packs 8 bf16 into one ds_store_b128.
__device__ __forceinline__ void stage_b_sw(const float* W, int srcStride, int c0,
                                           u16* l, int K, int NC) {
  int ntiles = NC >> 4;
  int total = K * NC;
  for (int base = threadIdx.x << 3; base < total; base += 256 * 8) {
    int lane = (base >> 4) & 31;
    int t = base >> 9;
    int kt = t / ntiles, nt = t - kt * ntiles;
    int half = lane >> 4;
    int n = c0 + (nt << 4) + (lane & 15);
    int e0 = base & 15;  // 0 or 8
    v8u v;
#pragma unroll
    for (int i = 0; i < 8; ++i) {
      int el = e0 + i;
      int k = (kt << 5) + ((el < 8) ? ((half << 3) + el) : (8 + (half << 3) + el));
      v[i] = f2bf(W[k * srcStride + n]);
    }
    *(v8u*)(l + base) = v;
  }
}

__device__ __forceinline__ v8f bias_frag(const float* b, int n0) {
  float bv = b[n0 + (threadIdx.x & 15)];
  v8f c;
#pragma unroll
  for (int i = 0; i < 8; ++i) c[i] = bv;
  return c;
}

// C/D: VGPR r holds row M = row0 + 8*(lane/16) + r, col N = n0 + lane%16
__device__ __forceinline__ void st_c(float* s, int row0, int n0, int stride, const v8f& c) {
  int lane = threadIdx.x & 31;
  int n = n0 + (lane & 15);
  int rbase = row0 + ((lane >> 4) << 3);
#pragma unroll
  for (int r = 0; r < 8; ++r) s[(rbase + r) * stride + n] = c[r];
}

// S[32][64] = A[32xK]@W[Kx64] + bias  (8 waves, one 16x16 tile each)
__device__ __forceinline__ void gemm32x64(const u16* A, int astr, const u16* W,
                                          const float* bias, float* S, int K) {
  int wv = threadIdx.x >> 5;
  int row0 = (wv & 1) << 4;
  int col0 = (wv >> 1) << 4;
  v8f acc = bias_frag(bias, col0);
  for (int k = 0; k < K; k += 32)
    acc = __builtin_amdgcn_wmma_f32_16x16x32_bf16(false, ld_a(A, row0, k, astr),
                                                  false, ld_b_sw(W, k >> 5, col0 >> 4, 4),
                                                  (short)0, acc, false, false);
  st_c(S, row0, col0, 64, acc);
}

// LayerNorm over 64 features held as 8 regs by each of 8 consecutive lanes
__device__ __forceinline__ void ln64(float v[8], const float* g, const float* b, int jb) {
  float s = 0.0f;
#pragma unroll
  for (int i = 0; i < 8; ++i) s += v[i];
#pragma unroll
  for (int m = 1; m < 8; m <<= 1) s += __shfl_xor(s, m, 32);
  float mean = s * 0.015625f;
  float q = 0.0f;
#pragma unroll
  for (int i = 0; i < 8; ++i) { float d = v[i] - mean; q += d * d; }
#pragma unroll
  for (int m = 1; m < 8; m <<= 1) q += __shfl_xor(q, m, 32);
  float inv = rsqrtf(q * 0.015625f + 1e-5f);
#pragma unroll
  for (int i = 0; i < 8; ++i) v[i] = (v[i] - mean) * inv * g[jb + i] + b[jb + i];
}

__device__ __forceinline__ void atomicMaxF(float* addr, float val) {
  if (val >= 0.0f) atomicMax((int*)addr, __float_as_int(val));
  else             atomicMin((unsigned int*)addr, (unsigned int)__float_as_int(val));
}

// pack 8 floats (regs) -> one b128 LDS store of bf16
__device__ __forceinline__ void pack8(u16* dst, const float v[8]) {
  v8u p;
#pragma unroll
  for (int i = 0; i < 8; ++i) p[i] = f2bf(v[i]);
  *(v8u*)dst = p;
}

// load 8 consecutive f32 from LDS (32B-aligned) into regs
__device__ __forceinline__ void lds8(const float* src, float v[8]) {
  v4f a = *(const v4f*)src;
  v4f b = *(const v4f*)(src + 4);
#pragma unroll
  for (int i = 0; i < 4; ++i) { v[i] = a[i]; v[4 + i] = b[i]; }
}

// stage node-feature tile (f32 global, clamped) -> bf16 LDS, one b128 store/thread
__device__ __forceinline__ void stage_a_node(const float* g, u16* l, int n0) {
  int tid = threadIdx.x;
  int r = tid >> 3, jb = (tid & 7) << 3;
  int nd = n0 + r;
  if (nd >= NN) nd = NN - 1;
  const float* src = g + (size_t)nd * 64 + jb;
  v4f a = *(const v4f*)src;
  v4f b = *(const v4f*)(src + 4);
  v8u v;
#pragma unroll
  for (int i = 0; i < 4; ++i) { v[i] = f2bf(a[i]); v[4 + i] = f2bf(b[i]); }
  *(v8u*)(l + r * 72 + jb) = v;
}

// ---------------- kernel 1: per-edge relative embedding (bf16 out) ----------
__global__ void __launch_bounds__(256)
k_rel_embed(const float* __restrict__ pos, const float* __restrict__ rotm,
            const float* __restrict__ rang, const int* __restrict__ ei,
            const float* __restrict__ P, const int* __restrict__ hsp,
            u16* __restrict__ relEmb) {
  __shared__ __align__(16) float geom[32][4];
  __shared__ __align__(16) u16 zt[32 * 72];
  __shared__ __align__(16) u16 wb[64 * 64];
  __shared__ __align__(16) float s[32 * 64];
  int tid = threadIdx.x;
  int e0 = blockIdx.x * 32;
  if (tid < 32) {
    int e = e0 + tid;
    int sn = ei[e], dn = ei[NE + e];
    int t = hsp[0] - 1;
    float px = pos[sn * 40 + t * 2 + 0] - pos[dn * 40 + t * 2 + 0];
    float py = pos[sn * 40 + t * 2 + 1] - pos[dn * 40 + t * 2 + 1];
    const float* M = rotm + dn * 4;
    geom[tid][0] = px * M[0] + py * M[2];
    geom[tid][1] = px * M[1] + py * M[3];
    float th = rang[sn] - rang[dn];
    geom[tid][2] = __cosf(th);
    geom[tid][3] = __sinf(th);
  }
  __syncthreads();
  int r = tid >> 3, jb = (tid & 7) << 3;
  int wv = tid >> 5, row0 = (wv & 1) << 4, col0 = (wv >> 1) << 4;
  v8f acc;
  {
    int n = col0 + (tid & 15);
    float bs = P[R_B0_L2_B + n] + P[R_B1_L2_B + n];
#pragma unroll
    for (int i = 0; i < 8; ++i) acc[i] = bs;
  }
  for (int br = 0; br < 2; ++br) {
    const float* W1 = P + (br ? R_B1_L1_W : R_B0_L1_W);
    const float* B1 = P + (br ? R_B1_L1_B : R_B0_L1_B);
    const float* G  = P + (br ? R_B1_N1_G : R_B0_N1_G);
    const float* Bn = P + (br ? R_B1_N1_B : R_B0_N1_B);
    const float* W2 = P + (br ? R_B1_L2_W : R_B0_L2_W);
    float in0 = geom[r][br * 2 + 0], in1 = geom[r][br * 2 + 1];
    float v[8];
#pragma unroll
    for (int i = 0; i < 8; ++i) {
      int j = jb + i;
      v[i] = in0 * W1[j] + in1 * W1[64 + j] + B1[j];
    }
    ln64(v, G, Bn, jb);
#pragma unroll
    for (int i = 0; i < 8; ++i) v[i] = fmaxf(v[i], 0.0f);
    pack8(zt + r * 72 + jb, v);
    stage_b_sw(W2, 64, 0, wb, 64, 64);
    __syncthreads();
    for (int k = 0; k < 64; k += 32)
      acc = __builtin_amdgcn_wmma_f32_16x16x32_bf16(false, ld_a(zt, row0, k, 72),
                                                    false, ld_b_sw(wb, k >> 5, col0 >> 4, 4),
                                                    (short)0, acc, false, false);
    __syncthreads();
  }
  st_c(s, row0, col0, 64, acc);
  __syncthreads();
  {
    float v[8];
    lds8(s + r * 64 + jb, v);
    ln64(v, P + R_AN1_G, P + R_AN1_B, jb);
#pragma unroll
    for (int i = 0; i < 8; ++i) v[i] = fmaxf(v[i], 0.0f);
    pack8(zt + r * 72 + jb, v);
  }
  stage_b_sw(P + R_AL_W, 64, 0, wb, 64, 64);
  __syncthreads();
  gemm32x64(zt, 72, wb, P + R_AL_B, s, 64);
  __syncthreads();
  {
    float v[8];
    lds8(s + r * 64 + jb, v);
    ln64(v, P + R_AN2_G, P + R_AN2_B, jb);
    v8u pv;
#pragma unroll
    for (int i = 0; i < 8; ++i) pv[i] = f2bf(v[i]);
    *(v8u*)(relEmb + (size_t)(e0 + r) * 64 + jb) = pv;
  }
}

// ---------------- kernel 2: node-side LN + q/kn/vn GEMMs --------------------
__device__ __forceinline__ void pre_gemm(const float* P, int wofs, int bofs,
                                         const u16* ht, u16* wb, float* s,
                                         float* outg, int node, int r, int jb) {
  __syncthreads();
  stage_b_sw(P + wofs, 64, 0, wb, 64, 64);
  __syncthreads();
  gemm32x64(ht, 72, wb, P + bofs, s, 64);
  __syncthreads();
  if (node < NN) {
    *(v4f*)(outg + (size_t)node * 64 + jb)     = *(const v4f*)(s + r * 64 + jb);
    *(v4f*)(outg + (size_t)node * 64 + jb + 4) = *(const v4f*)(s + r * 64 + jb + 4);
  }
}

__global__ void __launch_bounds__(256)
k_node_pre(const float* __restrict__ P, int lb, const float* __restrict__ x,
           float* __restrict__ h, float* __restrict__ q,
           float* __restrict__ kn, float* __restrict__ vn) {
  __shared__ __align__(16) u16 ht[32 * 72];
  __shared__ __align__(16) u16 wb[64 * 64];
  __shared__ __align__(16) float s[32 * 64];
  int tid = threadIdx.x;
  int n0 = blockIdx.x * 32;
  int r = tid >> 3, jb = (tid & 7) << 3;
  int node = n0 + r;
  int cn = node < NN ? node : NN - 1;
  float v[8];
  {
    v4f a = *(const v4f*)(x + (size_t)cn * 64 + jb);
    v4f b = *(const v4f*)(x + (size_t)cn * 64 + jb + 4);
#pragma unroll
    for (int i = 0; i < 4; ++i) { v[i] = a[i]; v[4 + i] = b[i]; }
  }
  ln64(v, P + lb + L_N1_G, P + lb + L_N1_B, jb);
  pack8(ht + r * 72 + jb, v);
  if (node < NN) {
    v4f a, b;
#pragma unroll
    for (int i = 0; i < 4; ++i) { a[i] = v[i]; b[i] = v[4 + i]; }
    *(v4f*)(h + (size_t)node * 64 + jb)     = a;
    *(v4f*)(h + (size_t)node * 64 + jb + 4) = b;
  }
  pre_gemm(P, lb + L_Q_W,  lb + L_Q_B,  ht, wb, s, q,  node, r, jb);
  pre_gemm(P, lb + L_KN_W, lb + L_KN_B, ht, wb, s, kn, node, r, jb);
  pre_gemm(P, lb + L_VN_W, lb + L_VN_B, ht, wb, s, vn, node, r, jb);
}

// ---------------- kernel 3: init segment buffers ----------------------------
__global__ void k_init(float* agg, float* nmax, float* nsum) {
  int i = blockIdx.x * 256 + threadIdx.x;
  if (i < NN * 64) agg[i] = 0.0f;
  if (i < NN * 8) { nmax[i] = -__builtin_inff(); nsum[i] = 0.0f; }
}

// ---------------- kernel 4: ke GEMM + alpha + segment max -------------------
__global__ void __launch_bounds__(256)
k_alpha(const float* __restrict__ P, int lb, const int* __restrict__ ei,
        const u16* __restrict__ relEmb, const float* __restrict__ q,
        const float* __restrict__ kn, float* __restrict__ alpha,
        float* __restrict__ nmax) {
  __shared__ __align__(16) u16 ret[32 * 72];
  __shared__ __align__(16) u16 wb[64 * 64];
  __shared__ __align__(16) float s[32 * 64];
  int tid = threadIdx.x;
  int e0 = blockIdx.x * 32;
  int r = tid >> 3;
  int jb8 = (tid & 7) << 3;
  __builtin_prefetch(relEmb + ((size_t)e0 + 32) * 64, 0, 1);
  *(v8u*)(ret + r * 72 + jb8) = *(const v8u*)(relEmb + ((size_t)e0 + r) * 64 + jb8);
  stage_b_sw(P + lb + L_KE_W, 64, 0, wb, 64, 64);
  __syncthreads();
  gemm32x64(ret, 72, wb, P + lb + L_KE_B, s, 64);
  __syncthreads();
  int hd = tid & 7;
  int e = e0 + r;
  int sn = ei[e], dn = ei[NE + e];
  float qv[8], kv[8], ke[8];
  {
    v4f a = *(const v4f*)(q + (size_t)dn * 64 + hd * 8);
    v4f b = *(const v4f*)(q + (size_t)dn * 64 + hd * 8 + 4);
    v4f c = *(const v4f*)(kn + (size_t)sn * 64 + hd * 8);
    v4f d = *(const v4f*)(kn + (size_t)sn * 64 + hd * 8 + 4);
#pragma unroll
    for (int i = 0; i < 4; ++i) { qv[i] = a[i]; qv[4 + i] = b[i]; kv[i] = c[i]; kv[4 + i] = d[i]; }
  }
  lds8(s + r * 64 + hd * 8, ke);
  float a = 0.0f;
#pragma unroll
  for (int d = 0; d < 8; ++d) a += qv[d] * (kv[d] + ke[d]);
  a *= 0.35355339059327373f;  // 1/sqrt(HEAD)
  alpha[(size_t)e * 8 + hd] = a;
  atomicMaxF(&nmax[dn * 8 + hd], a);
}

// ---------------- kernel 5: exp + segment sum -------------------------------
__global__ void __launch_bounds__(256)
k_norm(const int* __restrict__ ei, float* __restrict__ alpha,
       const float* __restrict__ nmax, float* __restrict__ nsum) {
  int i = blockIdx.x * 256 + threadIdx.x;  // < NE*8
  int e = i >> 3, hd = i & 7;
  int dn = ei[NE + e];
  float v = __expf(alpha[i] - nmax[dn * 8 + hd]);
  alpha[i] = v;
  atomicAdd(&nsum[dn * 8 + hd], v);
}

// ---------------- kernel 6: ve GEMM + weighted scatter-add ------------------
__global__ void __launch_bounds__(256)
k_agg(const float* __restrict__ P, int lb, const int* __restrict__ ei,
      const u16* __restrict__ relEmb, const float* __restrict__ vn,
      const float* __restrict__ alpha, const float* __restrict__ nsum,
      float* __restrict__ agg) {
  __shared__ __align__(16) u16 ret[32 * 72];
  __shared__ __align__(16) u16 wb[64 * 64];
  __shared__ __align__(16) float s[32 * 64];
  int tid = threadIdx.x;
  int e0 = blockIdx.x * 32;
  int r = tid >> 3;
  int jb = (tid & 7) << 3;
  __builtin_prefetch(relEmb + ((size_t)e0 + 32) * 64, 0, 1);
  *(v8u*)(ret + r * 72 + jb) = *(const v8u*)(relEmb + ((size_t)e0 + r) * 64 + jb);
  stage_b_sw(P + lb + L_VE_W, 64, 0, wb, 64, 64);
  __syncthreads();
  gemm32x64(ret, 72, wb, P + lb + L_VE_B, s, 64);
  __syncthreads();
  int hd = tid & 7;
  int e = e0 + r;
  int sn = ei[e], dn = ei[NE + e];
  float w = alpha[(size_t)e * 8 + hd] / (nsum[dn * 8 + hd] + 1e-16f);
  float vv[8], ve[8];
  {
    v4f a = *(const v4f*)(vn + (size_t)sn * 64 + jb);
    v4f b = *(const v4f*)(vn + (size_t)sn * 64 + jb + 4);
#pragma unroll
    for (int i = 0; i < 4; ++i) { vv[i] = a[i]; vv[4 + i] = b[i]; }
  }
  lds8(s + r * 64 + jb, ve);
#pragma unroll
  for (int i = 0; i < 8; ++i)
    atomicAdd(&agg[(size_t)dn * 64 + jb + i], (vv[i] + ve[i]) * w);
}

// ---------------- kernel 7: gated update + FFN ------------------------------
__global__ void __launch_bounds__(256)
k_node_upd(const float* __restrict__ P, int lb, const float* __restrict__ h,
           const float* __restrict__ agg, float* __restrict__ x) {
  __shared__ __align__(16) u16 ht[32 * 72];    // h bf16; later h2
  __shared__ __align__(16) u16 at[32 * 72];    // agg bf16; later upd
  __shared__ __align__(16) u16 ft[32 * 264];   // ffn hidden bf16
  __shared__ __align__(16) u16 wb[128 * 64];   // swizzled weight staging
  __shared__ __align__(16) float s1[32 * 64];
  __shared__ __align__(16) float s2[32 * 64];
  int tid = threadIdx.x;
  int n0 = blockIdx.x * 32;
  int r = tid >> 3, jb = (tid & 7) << 3;
  int node = n0 + r;
  int cn = node < NN ? node : NN - 1;
  int wv = tid >> 5, row0 = (wv & 1) << 4, col0 = (wv >> 1) << 4;
  stage_a_node(h, ht, n0);
  stage_a_node(agg, at, n0);
  // gate pre-activation: agg@Wih + h@Whh + b_ih + b_hh (double-accumulated WMMA)
  v8f acc;
  {
    int n = col0 + (tid & 15);
    float b = P[lb + L_IH_B + n] + P[lb + L_HH_B + n];
#pragma unroll
    for (int i = 0; i < 8; ++i) acc[i] = b;
  }
  stage_b_sw(P + lb + L_IH_W, 64, 0, wb, 64, 64);
  __syncthreads();
  for (int k = 0; k < 64; k += 32)
    acc = __builtin_amdgcn_wmma_f32_16x16x32_bf16(false, ld_a(at, row0, k, 72),
                                                  false, ld_b_sw(wb, k >> 5, col0 >> 4, 4),
                                                  (short)0, acc, false, false);
  __syncthreads();
  stage_b_sw(P + lb + L_HH_W, 64, 0, wb, 64, 64);
  __syncthreads();
  for (int k = 0; k < 64; k += 32)
    acc = __builtin_amdgcn_wmma_f32_16x16x32_bf16(false, ld_a(ht, row0, k, 72),
                                                  false, ld_b_sw(wb, k >> 5, col0 >> 4, 4),
                                                  (short)0, acc, false, false);
  st_c(s1, row0, col0, 64, acc);
  __syncthreads();
  stage_b_sw(P + lb + L_SELF_W, 64, 0, wb, 64, 64);
  __syncthreads();
  gemm32x64(ht, 72, wb, P + lb + L_SELF_B, s2, 64);
  __syncthreads();
  float xv[8];
  {
    float av[8], g1[8], g2[8];
    {
      v4f a = *(const v4f*)(agg + (size_t)cn * 64 + jb);
      v4f b = *(const v4f*)(agg + (size_t)cn * 64 + jb + 4);
      v4f c = *(const v4f*)(x + (size_t)cn * 64 + jb);
      v4f d = *(const v4f*)(x + (size_t)cn * 64 + jb + 4);
#pragma unroll
      for (int i = 0; i < 4; ++i) { av[i] = a[i]; av[4 + i] = b[i]; xv[i] = c[i]; xv[4 + i] = d[i]; }
    }
    lds8(s1 + r * 64 + jb, g1);
    lds8(s2 + r * 64 + jb, g2);
    float uv[8];
#pragma unroll
    for (int i = 0; i < 8; ++i) {
      float g = 1.0f / (1.0f + __expf(-g1[i]));
      uv[i] = av[i] + g * (g2[i] - av[i]);
    }
    pack8(at + r * 72 + jb, uv);
  }
  __syncthreads();
  stage_b_sw(P + lb + L_OUT_W, 64, 0, wb, 64, 64);
  __syncthreads();
  gemm32x64(at, 72, wb, P + lb + L_OUT_B, s1, 64);
  __syncthreads();
  {
    float o[8], v[8];
    lds8(s1 + r * 64 + jb, o);
#pragma unroll
    for (int i = 0; i < 8; ++i) { xv[i] += o[i]; v[i] = xv[i]; }
    ln64(v, P + lb + L_N2_G, P + lb + L_N2_B, jb);
    pack8(ht + r * 72 + jb, v);
  }
  __syncthreads();
  // FFN1: 64->256 in two 128-column halves (relu fused at fragment store)
  for (int half = 0; half < 2; ++half) {
    int c0 = half * 128;
    stage_b_sw(P + lb + L_FC1_W, 256, c0, wb, 64, 128);
    __syncthreads();
    for (int ct = 0; ct < 2; ++ct) {
      int cc = (wv >> 1) * 16 + ct * 64;
      v8f a2 = bias_frag(P + lb + L_FC1_B + c0, cc);
      for (int k = 0; k < 64; k += 32)
        a2 = __builtin_amdgcn_wmma_f32_16x16x32_bf16(false, ld_a(ht, row0, k, 72),
                                                     false, ld_b_sw(wb, k >> 5, cc >> 4, 8),
                                                     (short)0, a2, false, false);
      int lane = tid & 31;
      int n = c0 + cc + (lane & 15);
      int rbase = row0 + ((lane >> 4) << 3);
#pragma unroll
      for (int rr = 0; rr < 8; ++rr)
        ft[(rbase + rr) * 264 + n] = f2bf(fmaxf(a2[rr], 0.0f));
    }
    __syncthreads();
  }
  // FFN2: 256->64, K split in two 128 halves
  v8f a3 = bias_frag(P + lb + L_FC2_B, col0);
  for (int kh = 0; kh < 2; ++kh) {
    int k0 = kh * 128;
    stage_b_sw(P + lb + L_FC2_W + k0 * 64, 64, 0, wb, 128, 64);
    __syncthreads();
    for (int k = 0; k < 128; k += 32)
      a3 = __builtin_amdgcn_wmma_f32_16x16x32_bf16(false, ld_a(ft, row0, k0 + k, 264),
                                                   false, ld_b_sw(wb, k >> 5, col0 >> 4, 4),
                                                   (short)0, a3, false, false);
    __syncthreads();
  }
  st_c(s1, row0, col0, 64, a3);
  __syncthreads();
  if (node < NN) {
    float o[8];
    lds8(s1 + r * 64 + jb, o);
    v4f a, b;
#pragma unroll
    for (int i = 0; i < 4; ++i) { a[i] = xv[i] + o[i]; b[i] = xv[4 + i] + o[4 + i]; }
    *(v4f*)(x + (size_t)node * 64 + jb)     = a;
    *(v4f*)(x + (size_t)node * 64 + jb + 4) = b;
  }
}

// ---------------- kernel 8: final LN + 6-mode projection (transposed out) ---
__global__ void __launch_bounds__(256)
k_final(const float* __restrict__ P, const float* __restrict__ x,
        float* __restrict__ out) {
  __shared__ __align__(16) u16 ht[32 * 72];
  __shared__ __align__(16) u16 wB[64 * 384];
  int tid = threadIdx.x;
  int n0 = blockIdx.x * 32;
  int r = tid >> 3, jb = (tid & 7) << 3;
  int node = n0 + r;
  int cn = node < NN ? node : NN - 1;
  float v[8];
  {
    v4f a = *(const v4f*)(x + (size_t)cn * 64 + jb);
    v4f b = *(const v4f*)(x + (size_t)cn * 64 + jb + 4);
#pragma unroll
    for (int i = 0; i < 4; ++i) { v[i] = a[i]; v[4 + i] = b[i]; }
  }
  ln64(v, P + P_NORM_G, P + P_NORM_B, jb);
  pack8(ht + r * 72 + jb, v);
  stage_b_sw(P + P_PROJ_W, 384, 0, wB, 64, 384);
  __syncthreads();
  int wv = tid >> 5, row0 = (wv & 1) << 4;
  int lane = tid & 31;
  for (int ct = 0; ct < 6; ++ct) {
    int c0 = (wv >> 1) * 16 + ct * 64;
    v8f acc = bias_frag(P + P_PROJ_B, c0);
    for (int k = 0; k < 64; k += 32)
      acc = __builtin_amdgcn_wmma_f32_16x16x32_bf16(false, ld_a(ht, row0, k, 72),
                                                    false, ld_b_sw(wB, k >> 5, c0 >> 4, 24),
                                                    (short)0, acc, false, false);
    int col = c0 + (lane & 15);
    int mode = col >> 6, d = col & 63;
    int rbase = row0 + ((lane >> 4) << 3);
#pragma unroll
    for (int rr = 0; rr < 8; ++rr) {
      int nd = n0 + rbase + rr;
      if (nd < NN) out[(size_t)mode * NN * 64 + (size_t)nd * 64 + d] = acc[rr];
    }
  }
}

// ---------------- host orchestration ----------------------------------------
extern "C" void kernel_launch(void* const* d_in, const int* in_sizes, int n_in,
                              void* d_out, int out_size, void* d_ws, size_t ws_size,
                              hipStream_t stream) {
  const float* pos  = (const float*)d_in[1];
  const float* rotm = (const float*)d_in[2];
  const float* rang = (const float*)d_in[3];
  const float* P    = (const float*)d_in[4];
  const int*   ei   = (const int*)d_in[5];
  const int*   hsp  = (const int*)d_in[6];
  float* out = (float*)d_out;

  char* w = (char*)d_ws;
  u16*   relEmb = (u16*)w;   w += (size_t)NE * 64 * sizeof(u16);   // 128 MB (bf16)
  float* alpha  = (float*)w; w += (size_t)NE * 8 * sizeof(float);  // 32 MB
  float* xbuf   = (float*)w; w += (size_t)NN * 64 * sizeof(float);
  float* hbuf   = (float*)w; w += (size_t)NN * 64 * sizeof(float);
  float* qbuf   = (float*)w; w += (size_t)NN * 64 * sizeof(float);
  float* knbuf  = (float*)w; w += (size_t)NN * 64 * sizeof(float);
  float* vnbuf  = (float*)w; w += (size_t)NN * 64 * sizeof(float);
  float* aggbuf = (float*)w; w += (size_t)NN * 64 * sizeof(float);
  float* nmax   = (float*)w; w += (size_t)NN * 8 * sizeof(float);
  float* nsum   = (float*)w; w += (size_t)NN * 8 * sizeof(float);

  hipMemcpyAsync(xbuf, d_in[0], (size_t)NN * 64 * sizeof(float),
                 hipMemcpyDeviceToDevice, stream);

  dim3 blk(256);
  int egrid = NE / 32;            // 31250
  int ngrid = (NN + 31) / 32;     // 1563

  k_rel_embed<<<egrid, blk, 0, stream>>>(pos, rotm, rang, ei, P, hsp, relEmb);

  for (int l = 0; l < NLAYER; ++l) {
    int lb = l * LAYER_SZ;
    k_node_pre<<<ngrid, blk, 0, stream>>>(P, lb, xbuf, hbuf, qbuf, knbuf, vnbuf);
    k_init<<<(NN * 64 + 255) / 256, blk, 0, stream>>>(aggbuf, nmax, nsum);
    k_alpha<<<egrid, blk, 0, stream>>>(P, lb, ei, relEmb, qbuf, knbuf, alpha, nmax);
    k_norm<<<(NE * 8) / 256, blk, 0, stream>>>(ei, alpha, nmax, nsum);
    k_agg<<<egrid, blk, 0, stream>>>(P, lb, ei, relEmb, vnbuf, alpha, nsum, aggbuf);
    k_node_upd<<<ngrid, blk, 0, stream>>>(P, lb, hbuf, aggbuf, xbuf);
  }
  k_final<<<ngrid, blk, 0, stream>>>(P, xbuf, out);
}
